// RoPESelfAttention_27771258536724
// MI455X (gfx1250) — compile-verified, matrix operand order checked by reading, not run on hardware
//
#include <hip/hip_runtime.h>
#include <hip/hip_bf16.h>
#include <stdint.h>

// Problem constants (match reference)
#define D_MODEL 1024
#define NHEAD   16
#define DK      64
#define BB      2
#define LSEQ    2048

typedef __attribute__((ext_vector_type(16))) __bf16 v16bf;
typedef __attribute__((ext_vector_type(8)))  float  v8f;

#define WMMA_BF16(a,b,c) \
  __builtin_amdgcn_wmma_f32_16x16x32_bf16(false,(a),false,(b),(short)0,(c),false,false)

__device__ __forceinline__ v8f zero8() {
  v8f z;
#pragma unroll
  for (int i = 0; i < 8; ++i) z[i] = 0.0f;
  return z;
}

__device__ __forceinline__ unsigned short f2bf(float f) {
  unsigned u = __float_as_uint(f);
  return (unsigned short)((u + 0x8000u) >> 16);  // round-to-nearest (approx)
}

union FragU { uint4 q[2]; v16bf v; };

// Load one 16x32 bf16 A/B fragment per the CDNA5 VGPR layout:
// lane half sel==0: elems0-7 = K[kOff+0..7],  elems8-15 = K[kOff+16..23]
// lane half sel==1: elems0-7 = K[kOff+8..15], elems8-15 = K[kOff+24..31]
__device__ __forceinline__ v16bf load_frag(const unsigned short* base, int row,
                                           int stride, int kOff, int sel) {
  const unsigned short* p = base + row * stride + kOff + (sel << 3);
  FragU f;
  f.q[0] = *(const uint4*)(p);
  f.q[1] = *(const uint4*)(p + 16);
  return f.v;
}

// stage 32 consecutive fp32 -> 32 bf16 (VALU convert path)
__device__ __forceinline__ void stage_f32(const float* __restrict__ src,
                                          unsigned short* dst) {
#pragma unroll
  for (int i = 0; i < 8; ++i) {
    float4 v = ((const float4*)src)[i];
    ushort4 o;
    o.x = f2bf(v.x); o.y = f2bf(v.y); o.z = f2bf(v.z); o.w = f2bf(v.w);
    ((ushort4*)dst)[i] = o;
  }
}

// ---- CDNA5 async global->LDS copies (tracked by ASYNCcnt) -----------------
// Per-lane: copy 64 contiguous bytes global -> LDS. INST_OFFSET applies to
// both the LDS and global address, so one address setup covers 4 x B128.
__device__ __forceinline__ void async_copy64B(const unsigned short* g,
                                              unsigned short* l) {
  unsigned lds = (unsigned)(unsigned long long)l;  // low 32b = LDS offset
  asm volatile(
      "global_load_async_to_lds_b128 %0, %1, off\n\t"
      "global_load_async_to_lds_b128 %0, %1, off offset:16\n\t"
      "global_load_async_to_lds_b128 %0, %1, off offset:32\n\t"
      "global_load_async_to_lds_b128 %0, %1, off offset:48"
      :: "v"(lds), "v"(g) : "memory");
}
__device__ __forceinline__ void async_copy128B(const unsigned short* g,
                                               unsigned short* l) {
  unsigned lds = (unsigned)(unsigned long long)l;
  asm volatile(
      "global_load_async_to_lds_b128 %0, %1, off\n\t"
      "global_load_async_to_lds_b128 %0, %1, off offset:16\n\t"
      "global_load_async_to_lds_b128 %0, %1, off offset:32\n\t"
      "global_load_async_to_lds_b128 %0, %1, off offset:48\n\t"
      "global_load_async_to_lds_b128 %0, %1, off offset:64\n\t"
      "global_load_async_to_lds_b128 %0, %1, off offset:80\n\t"
      "global_load_async_to_lds_b128 %0, %1, off offset:96\n\t"
      "global_load_async_to_lds_b128 %0, %1, off offset:112"
      :: "v"(lds), "v"(g) : "memory");
}
__device__ __forceinline__ void wait_async0() {
  asm volatile("s_wait_asynccnt 0x0" ::: "memory");
}

#define LDS_STRIDE 72  // 64 + 8 bf16 pad

// ---------------------------------------------------------------------------
// Kernel 1: qkv = x @ W_qkv^T + b_qkv, apply RoPE to q,k, scatter bf16
//   Qr, Kr : [B*H][L][64] bf16   Vt : [B*H][64][L] bf16 (transposed for B-frag)
// grid (48, 32), block 128 (4 waves); tile 128(M) x 64(N), K chunk 64.
// Each wave computes 32 rows x 64 cols (2 M-frags) for better A/B reuse.
// ---------------------------------------------------------------------------
__global__ __launch_bounds__(128) void qkv_rope_kernel(
    const float* __restrict__ x, const float* __restrict__ rope_cos,
    const float* __restrict__ rope_sin, const float* __restrict__ W_qkv,
    const float* __restrict__ b_qkv, unsigned short* __restrict__ Qr,
    unsigned short* __restrict__ Kr, unsigned short* __restrict__ Vt) {
  __shared__ unsigned short aL[128 * LDS_STRIDE];
  __shared__ unsigned short bL[64 * LDS_STRIDE];

  const int t = threadIdx.x;
  const int w = t >> 5, lane = t & 31;
  const int lhi = lane >> 4, llo = lane & 15;
  const int n0 = blockIdx.x * 64;
  const int m0 = blockIdx.y * 128;

  v8f acc[2][4];
#pragma unroll
  for (int mf = 0; mf < 2; ++mf)
#pragma unroll
    for (int f = 0; f < 4; ++f) acc[mf][f] = zero8();

  for (int k0 = 0; k0 < D_MODEL; k0 += 64) {
    __syncthreads();
    {  // A tile: 128 rows x 64 cols, thread t stages row t (fp32 -> bf16)
      const float* src = x + (size_t)(m0 + t) * D_MODEL + k0;
      unsigned short* dst = aL + t * LDS_STRIDE;
      stage_f32(src, dst);
      stage_f32(src + 32, dst + 32);
    }
    {  // B tile: 64 rows x 64 cols, thread t stages half a row
      stage_f32(W_qkv + (size_t)(n0 + (t >> 1)) * D_MODEL + k0 + (t & 1) * 32,
                bL + (t >> 1) * LDS_STRIDE + (t & 1) * 32);
    }
    __syncthreads();
#pragma unroll
    for (int ks = 0; ks < 64; ks += 32) {
      v16bf a0 = load_frag(aL, w * 32 + llo, LDS_STRIDE, ks, lhi);
      v16bf a1 = load_frag(aL, w * 32 + 16 + llo, LDS_STRIDE, ks, lhi);
#pragma unroll
      for (int f = 0; f < 4; ++f) {
        v16bf b = load_frag(bL, f * 16 + llo, LDS_STRIDE, ks, lhi);
        acc[0][f] = WMMA_BF16(a0, b, acc[0][f]);
        acc[1][f] = WMMA_BF16(a1, b, acc[1][f]);
      }
    }
  }

  // Epilogue: bias + RoPE + scatter
  const int which = n0 >> 10;            // 0=q 1=k 2=v
  const int h = (n0 & 1023) >> 6;        // head
#pragma unroll
  for (int mf = 0; mf < 2; ++mf) {
    const int mbase = m0 + w * 32 + mf * 16 + 8 * lhi;
    float val[4][8];
#pragma unroll
    for (int f = 0; f < 4; ++f) {
      float bias = b_qkv[n0 + f * 16 + llo];
#pragma unroll
      for (int rr = 0; rr < 8; ++rr) val[f][rr] = acc[mf][f][rr] + bias;
    }

    if (which < 2) {
#pragma unroll
      for (int f = 0; f < 2; ++f) {
        int d = f * 16 + llo;  // 0..31
#pragma unroll
        for (int rr = 0; rr < 8; ++rr) {
          int m = mbase + rr;
          float c = rope_cos[m * (DK / 2) + d];
          float s = rope_sin[m * (DK / 2) + d];
          float x1 = val[f][rr], x2 = val[f + 2][rr];
          val[f][rr]     = x1 * c - x2 * s;
          val[f + 2][rr] = x1 * s + x2 * c;
        }
      }
      unsigned short* dst = (which == 0) ? Qr : Kr;
#pragma unroll
      for (int f = 0; f < 4; ++f) {
        int d = f * 16 + llo;
#pragma unroll
        for (int rr = 0; rr < 8; ++rr) {
          int m = mbase + rr;
          int b = m >> 11, pos = m & (LSEQ - 1);
          dst[((size_t)(b * NHEAD + h) * LSEQ + pos) * DK + d] = f2bf(val[f][rr]);
        }
      }
    } else {
#pragma unroll
      for (int f = 0; f < 4; ++f) {
        int d = f * 16 + llo;
#pragma unroll
        for (int rr = 0; rr < 8; ++rr) {
          int m = mbase + rr;
          int b = m >> 11, pos = m & (LSEQ - 1);
          Vt[((size_t)(b * NHEAD + h) * DK + d) * LSEQ + pos] = f2bf(val[f][rr]);
        }
      }
    }
  }
}

// ---------------------------------------------------------------------------
// Kernel 2: flash attention per head.  grid (32, 32) = (q-tiles, B*H), block 128.
// Each wave: 16 query rows, online softmax, O written as [B][L][H*64] bf16.
// K/V tiles staged with CDNA5 async global->LDS copies (ASYNCcnt).
// ---------------------------------------------------------------------------
__global__ __launch_bounds__(128) void attn_kernel(
    const unsigned short* __restrict__ Qr, const unsigned short* __restrict__ Kr,
    const unsigned short* __restrict__ Vt, unsigned short* __restrict__ O) {
  __shared__ unsigned short kL[64 * LDS_STRIDE];
  __shared__ unsigned short vL[64 * LDS_STRIDE];
  __shared__ unsigned short pL[4][16 * LDS_STRIDE];

  const int t = threadIdx.x;
  const int w = t >> 5, lane = t & 31;
  const int lhi = lane >> 4, llo = lane & 15;
  const int bh = blockIdx.y;                   // 0..31 = b*16+h
  const int q0 = blockIdx.x * 64 + w * 16;     // wave's first query row
  const size_t hb = (size_t)bh * LSEQ * DK;    // same element offset for Qr/Kr/Vt

  // Q fragments (held in registers for whole kernel)
  v16bf aQ[2];
#pragma unroll
  for (int c = 0; c < 2; ++c)
    aQ[c] = load_frag(Qr + hb, q0 + llo, DK, c * 32, lhi);

  v8f acc[4];
#pragma unroll
  for (int f = 0; f < 4; ++f) acc[f] = zero8();
  float mrun[8], lrun[8];
#pragma unroll
  for (int rr = 0; rr < 8; ++rr) { mrun[rr] = -1e30f; lrun[rr] = 0.0f; }

  const int sr = t >> 1, sh = t & 1;
  const float scale = 0.125f;  // 1/sqrt(64)

  for (int kt = 0; kt < LSEQ; kt += 64) {
    __syncthreads();
    // K tile (rows = key pos) and V tile (rows = d; Vt pre-transposed):
    // async DMA, 64B per thread each, no VGPR round-trip.
    async_copy64B(Kr + hb + (size_t)(kt + sr) * DK + sh * 32,
                  kL + sr * LDS_STRIDE + sh * 32);
    async_copy64B(Vt + hb + (size_t)sr * LSEQ + kt + sh * 32,
                  vL + sr * LDS_STRIDE + sh * 32);
    wait_async0();
    __syncthreads();

    // S = Q K^T
    v8f s[4];
#pragma unroll
    for (int f = 0; f < 4; ++f) s[f] = zero8();
#pragma unroll
    for (int c = 0; c < 2; ++c) {
#pragma unroll
      for (int f = 0; f < 4; ++f) {
        v16bf bK = load_frag(kL, f * 16 + llo, LDS_STRIDE, c * 32, lhi);
        s[f] = WMMA_BF16(aQ[c], bK, s[f]);
      }
    }

    // online softmax (row = VGPR index rr; 16 cols of a row live in one lane-half)
#pragma unroll
    for (int f = 0; f < 4; ++f)
#pragma unroll
      for (int rr = 0; rr < 8; ++rr) s[f][rr] *= scale;

#pragma unroll
    for (int rr = 0; rr < 8; ++rr) {
      float m = fmaxf(fmaxf(s[0][rr], s[1][rr]), fmaxf(s[2][rr], s[3][rr]));
      m = fmaxf(m, __shfl_xor(m, 1));
      m = fmaxf(m, __shfl_xor(m, 2));
      m = fmaxf(m, __shfl_xor(m, 4));
      m = fmaxf(m, __shfl_xor(m, 8));
      float mnew  = fmaxf(mrun[rr], m);
      float alpha = __expf(mrun[rr] - mnew);
      float rs = 0.0f;
#pragma unroll
      for (int f = 0; f < 4; ++f) {
        float p = __expf(s[f][rr] - mnew);
        s[f][rr] = p;
        rs += p;
      }
      rs += __shfl_xor(rs, 1);
      rs += __shfl_xor(rs, 2);
      rs += __shfl_xor(rs, 4);
      rs += __shfl_xor(rs, 8);
      lrun[rr] = lrun[rr] * alpha + rs;
      mrun[rr] = mnew;
#pragma unroll
      for (int f = 0; f < 4; ++f) acc[f][rr] *= alpha;
    }

    // P: D-layout -> LDS -> A-layout (per-wave private tile; same-wave DS order)
    unsigned short* pw = pL[w];
#pragma unroll
    for (int f = 0; f < 4; ++f)
#pragma unroll
      for (int rr = 0; rr < 8; ++rr)
        pw[(rr + 8 * lhi) * LDS_STRIDE + f * 16 + llo] = f2bf(s[f][rr]);
    asm volatile("s_wait_dscnt 0" ::: "memory");

    // O += P V
#pragma unroll
    for (int c = 0; c < 2; ++c) {
      v16bf aP = load_frag(pw, llo, LDS_STRIDE, c * 32, lhi);
#pragma unroll
      for (int f = 0; f < 4; ++f) {
        v16bf bV = load_frag(vL, f * 16 + llo, LDS_STRIDE, c * 32, lhi);
        acc[f] = WMMA_BF16(aP, bV, acc[f]);
      }
    }
  }

  // normalize + store O as [B][L][H*DK] bf16
  const int b = bh >> 4, h = bh & 15;
  float inv[8];
#pragma unroll
  for (int rr = 0; rr < 8; ++rr) inv[rr] = 1.0f / lrun[rr];
#pragma unroll
  for (int f = 0; f < 4; ++f) {
    int d = f * 16 + llo;
#pragma unroll
    for (int rr = 0; rr < 8; ++rr) {
      int pos = q0 + rr + 8 * lhi;
      O[(size_t)(b * LSEQ + pos) * D_MODEL + h * DK + d] = f2bf(acc[f][rr] * inv[rr]);
    }
  }
}

// ---------------------------------------------------------------------------
// Kernel 3: out = O @ W_out^T + b_out (fp32 result). grid (16, 32), block 128.
// Tile 128(M) x 64(N); A tile (bf16) staged with async global->LDS DMA.
// ---------------------------------------------------------------------------
__global__ __launch_bounds__(128) void out_proj_kernel(
    const unsigned short* __restrict__ O, const float* __restrict__ W_out,
    const float* __restrict__ b_out, float* __restrict__ out) {
  __shared__ unsigned short aL[128 * LDS_STRIDE];
  __shared__ unsigned short bL[64 * LDS_STRIDE];

  const int t = threadIdx.x;
  const int w = t >> 5, lane = t & 31;
  const int lhi = lane >> 4, llo = lane & 15;
  const int n0 = blockIdx.x * 64;
  const int m0 = blockIdx.y * 128;

  v8f acc[2][4];
#pragma unroll
  for (int mf = 0; mf < 2; ++mf)
#pragma unroll
    for (int f = 0; f < 4; ++f) acc[mf][f] = zero8();

  for (int k0 = 0; k0 < D_MODEL; k0 += 64) {
    __syncthreads();
    // A tile: thread t stages row t (128 bytes) via async DMA
    async_copy128B(O + (size_t)(m0 + t) * D_MODEL + k0, aL + t * LDS_STRIDE);
    // B tile: fp32 -> bf16 convert path
    stage_f32(W_out + (size_t)(n0 + (t >> 1)) * D_MODEL + k0 + (t & 1) * 32,
              bL + (t >> 1) * LDS_STRIDE + (t & 1) * 32);
    wait_async0();
    __syncthreads();
#pragma unroll
    for (int ks = 0; ks < 64; ks += 32) {
      v16bf a0 = load_frag(aL, w * 32 + llo, LDS_STRIDE, ks, lhi);
      v16bf a1 = load_frag(aL, w * 32 + 16 + llo, LDS_STRIDE, ks, lhi);
#pragma unroll
      for (int f = 0; f < 4; ++f) {
        v16bf b = load_frag(bL, f * 16 + llo, LDS_STRIDE, ks, lhi);
        acc[0][f] = WMMA_BF16(a0, b, acc[0][f]);
        acc[1][f] = WMMA_BF16(a1, b, acc[1][f]);
      }
    }
  }

#pragma unroll
  for (int mf = 0; mf < 2; ++mf) {
    const int mbase = m0 + w * 32 + mf * 16 + 8 * lhi;
#pragma unroll
    for (int f = 0; f < 4; ++f) {
      int c = n0 + f * 16 + llo;
      float bias = b_out[c];
#pragma unroll
      for (int rr = 0; rr < 8; ++rr)
        out[(size_t)(mbase + rr) * D_MODEL + c] = acc[mf][f][rr] + bias;
    }
  }
}

// ---------------------------------------------------------------------------
extern "C" void kernel_launch(void* const* d_in, const int* in_sizes, int n_in,
                              void* d_out, int out_size, void* d_ws, size_t ws_size,
                              hipStream_t stream) {
  (void)in_sizes; (void)n_in; (void)out_size; (void)ws_size;
  const float* x        = (const float*)d_in[0];
  const float* rope_cos = (const float*)d_in[1];
  const float* rope_sin = (const float*)d_in[2];
  const float* W_qkv    = (const float*)d_in[3];
  const float* b_qkv    = (const float*)d_in[4];
  const float* W_out    = (const float*)d_in[5];
  const float* b_out    = (const float*)d_in[6];

  const size_t headElems = (size_t)BB * NHEAD * LSEQ * DK;  // 4M elems each
  unsigned short* Qr = (unsigned short*)d_ws;
  unsigned short* Kr = Qr + headElems;
  unsigned short* Vt = Kr + headElems;
  unsigned short* O  = Vt + headElems;  // [B][L][D_MODEL] bf16

  qkv_rope_kernel<<<dim3(3 * D_MODEL / 64, BB * LSEQ / 128), 128, 0, stream>>>(
      x, rope_cos, rope_sin, W_qkv, b_qkv, Qr, Kr, Vt);
  attn_kernel<<<dim3(LSEQ / 64, BB * NHEAD), 128, 0, stream>>>(Qr, Kr, Vt, O);
  out_proj_kernel<<<dim3(D_MODEL / 64, BB * LSEQ / 128), 128, 0, stream>>>(
      O, W_out, b_out, (float*)d_out);
}
